// Qwen3GatedDeltaNet_12489764897583
// MI455X (gfx1250) — matrix operand order repaired and checked by reading, not compiled, not run on hardware
//
#include <hip/hip_runtime.h>

typedef unsigned short u16;
typedef unsigned int   u32;
typedef __attribute__((ext_vector_type(16))) __bf16 v16bf;
typedef __attribute__((ext_vector_type(8)))  float  v8f;
typedef __attribute__((ext_vector_type(8)))  u32    v8u;

// ---------------- problem constants ----------------
#define SEQ    4096
#define HID    2048
#define NQKVZ  12288   // 2*KEY_DIM + 2*VALUE_DIM
#define CONVD  8192    // 2*KEY_DIM + VALUE_DIM
#define NH16   16
#define NH32   32
#define DK     128
#define DV     128
#define NCHUNK 64
#define CHUNK  64

// ---------------- bf16 helpers ----------------
__device__ __forceinline__ u16 f2bf(float f) {
  u32 u = __float_as_uint(f);
  u32 r = (u + 0x7FFFu + ((u >> 16) & 1u)) >> 16;
  return (u16)r;
}
__device__ __forceinline__ float bf2f(u16 b) {
  return __uint_as_float(((u32)b) << 16);
}
__device__ __forceinline__ u32 pack2(float a, float b) {
  return (u32)f2bf(a) | ((u32)f2bf(b) << 16);
}
__device__ __forceinline__ v8f vzero() {
  v8f z = {0.f, 0.f, 0.f, 0.f, 0.f, 0.f, 0.f, 0.f};
  return z;
}
__device__ __forceinline__ v8f wmma_bf16(v16bf a, v16bf b, v8f c) {
  return __builtin_amdgcn_wmma_f32_16x16x32_bf16(false, a, false, b, (short)0, c,
                                                 false, false);
}

// ---------------- CDNA5 async global->LDS (ASYNCcnt path) ----------------
// Moves 16 bytes/lane memory->LDS without touching VGPR data paths.
__device__ __forceinline__ void async_g2l_b128(void* lds, const void* gaddr) {
  u32 loff = (u32)(size_t)lds;  // low 32 bits of generic ptr = LDS aperture offset
  asm volatile("global_load_async_to_lds_b128 %0, %1, off"
               :: "v"(loff), "v"(gaddr)
               : "memory");
}
__device__ __forceinline__ void wait_async0() {
  asm volatile("s_wait_asynccnt 0" ::: "memory");
}

// ---------------- WMMA fragment loaders (ISA 7.12.2 layouts, wave32) ----------------
// A: 16x32 bf16 row-major, lane m = lane&15 holds row m; K chunks at o and o+16
__device__ __forceinline__ v16bf load_a_bf16(const u16* base, int ld) {
  int lane = threadIdx.x & 31;
  const u16* p = base + (lane & 15) * ld + ((lane < 16) ? 0 : 8);
  uint4 q0 = *(const uint4*)(p);
  uint4 q1 = *(const uint4*)(p + 16);
  v8u raw;
  raw[0] = q0.x; raw[1] = q0.y; raw[2] = q0.z; raw[3] = q0.w;
  raw[4] = q1.x; raw[5] = q1.y; raw[6] = q1.z; raw[7] = q1.w;
  return __builtin_bit_cast(v16bf, raw);
}
// A from bf16 row-major with per-row scale (row = lane&15 within tile)
__device__ __forceinline__ v16bf load_a_bf16_scaled(const u16* base, int ld, float sc) {
  int lane = threadIdx.x & 31;
  const u16* p = base + (lane & 15) * ld + ((lane < 16) ? 0 : 8);
  v8u raw;
#pragma unroll
  for (int j = 0; j < 4; ++j) {
    raw[j]     = pack2(bf2f(p[2*j]) * sc,      bf2f(p[2*j+1]) * sc);
    raw[j + 4] = pack2(bf2f(p[16+2*j]) * sc,   bf2f(p[16+2*j+1]) * sc);
  }
  return __builtin_bit_cast(v16bf, raw);
}
// A from f32 row-major (converted to bf16 on the fly)
__device__ __forceinline__ v16bf load_a_f32(const float* base, int ld) {
  int lane = threadIdx.x & 31;
  const float* p = base + (lane & 15) * ld + ((lane < 16) ? 0 : 8);
  v8u raw;
#pragma unroll
  for (int j = 0; j < 4; ++j) {
    raw[j]     = pack2(p[2*j],      p[2*j+1]);
    raw[j + 4] = pack2(p[16+2*j],   p[16+2*j+1]);
  }
  return __builtin_bit_cast(v16bf, raw);
}
// B: 32x16 bf16 accessed via n-major (B^T) storage: base[n*ld + k]
__device__ __forceinline__ v16bf load_b_nmajor(const u16* base, int ld) {
  int lane = threadIdx.x & 31;
  const u16* p = base + (lane & 15) * ld + ((lane < 16) ? 0 : 8);
  uint4 q0 = *(const uint4*)(p);
  uint4 q1 = *(const uint4*)(p + 16);
  v8u raw;
  raw[0] = q0.x; raw[1] = q0.y; raw[2] = q0.z; raw[3] = q0.w;
  raw[4] = q1.x; raw[5] = q1.y; raw[6] = q1.z; raw[7] = q1.w;
  return __builtin_bit_cast(v16bf, raw);
}

// ---------------- K0a: f32 -> bf16 conversion ----------------
__global__ void cvt_bf16_kernel(const float* __restrict__ in, u16* __restrict__ out,
                                long long n) {
  long long i = (long long)blockIdx.x * blockDim.x + threadIdx.x;
  long long stride = (long long)gridDim.x * blockDim.x;
  for (; i < n; i += stride) out[i] = f2bf(in[i]);
}

// ---------------- K0b: f32 (KxN) -> bf16 transposed (NxK) ----------------
__global__ void __launch_bounds__(256)
transpose_cvt_kernel(const float* __restrict__ in, u16* __restrict__ out, int K, int N) {
  __shared__ float tile[32][33];
  int n0 = blockIdx.x * 32, k0 = blockIdx.y * 32;
  int tx = threadIdx.x & 31, ty = threadIdx.x >> 5;  // 32 x 8
#pragma unroll
  for (int i = ty; i < 32; i += 8)
    tile[i][tx] = in[(size_t)(k0 + i) * N + n0 + tx];
  __syncthreads();
#pragma unroll
  for (int i = ty; i < 32; i += 8)
    out[(size_t)(n0 + i) * K + k0 + tx] = f2bf(tile[tx][i]);
}

// ---------------- K1: bf16 WMMA GEMM, C f32 = A(MxK) * BT(NxK)^T ----------------
// Double-buffered LDS staging via GLOBAL_LOAD_ASYNC_TO_LDS_B128, overlapped with WMMA.
#define TM 128
#define TN 128
#define TK 32
__global__ void __launch_bounds__(256)
gemm_bf16(const u16* __restrict__ A, const u16* __restrict__ BT, float* __restrict__ C,
          int M, int N, int K) {
  __shared__ uint4 AsRaw[2][TM * TK / 8];  // 2 x 8 KB
  __shared__ uint4 BsRaw[2][TN * TK / 8];  // 2 x 8 KB (n-major tiles)

  int bn = blockIdx.x, bm = blockIdx.y;
  int row0 = bm * TM, col0 = bn * TN;
  int tid = threadIdx.x;
  int wave = tid >> 5;
  int wr = (wave >> 2) * 64;   // 0 / 64
  int wc = (wave & 3) * 32;    // 0,32,64,96

  // staging geometry: 256 threads x 32B -> one 128x32 bf16 tile per matrix
  int sr = (tid * 16) >> 5;        // tile row (0..127)
  int sc = (tid * 16) & 31;        // k offset (0 or 16)

  v8f acc[4][2];
#pragma unroll
  for (int i = 0; i < 4; ++i)
#pragma unroll
    for (int j = 0; j < 2; ++j) acc[i][j] = vzero();

  auto stage = [&](int buf, int k0) {
    u16* Ad = (u16*)AsRaw[buf] + sr * TK + sc;
    const u16* Asrc = A + (size_t)(row0 + sr) * K + k0 + sc;
    async_g2l_b128(Ad, Asrc);
    async_g2l_b128(Ad + 8, Asrc + 8);
    u16* Bd = (u16*)BsRaw[buf] + sr * TK + sc;
    const u16* Bsrc = BT + (size_t)(col0 + sr) * K + k0 + sc;
    async_g2l_b128(Bd, Bsrc);
    async_g2l_b128(Bd + 8, Bsrc + 8);
  };

  stage(0, 0);
  wait_async0();
  __syncthreads();

  int cur = 0;
  for (int k0 = 0; k0 < K; k0 += TK) {
    if (k0 + TK < K) stage(cur ^ 1, k0 + TK);          // async DMA of next tile
    if (k0 + 2 * TK < K) {                              // L2 warm for tile after that
      __builtin_prefetch(A + (size_t)(row0 + (tid & 127)) * K + k0 + 2 * TK, 0, 1);
      __builtin_prefetch(BT + (size_t)(col0 + (tid & 127)) * K + k0 + 2 * TK, 0, 1);
    }
    const u16* As  = (const u16*)AsRaw[cur];
    const u16* BsT = (const u16*)BsRaw[cur];
    v16bf bfrag[2];
#pragma unroll
    for (int j = 0; j < 2; ++j) bfrag[j] = load_b_nmajor(&BsT[(wc + j * 16) * TK], TK);
#pragma unroll
    for (int i = 0; i < 4; ++i) {
      v16bf afrag = load_a_bf16(&As[(wr + i * 16) * TK], TK);
#pragma unroll
      for (int j = 0; j < 2; ++j) acc[i][j] = wmma_bf16(afrag, bfrag[j], acc[i][j]);
    }
    wait_async0();     // next tile landed in LDS
    __syncthreads();
    cur ^= 1;
  }
  // store (C layout: VGPR r -> row r / r+8 by lane half; col = lane&15)
  int lane = tid & 31;
  int nn = lane & 15, mo = (lane < 16) ? 0 : 8;
#pragma unroll
  for (int i = 0; i < 4; ++i)
#pragma unroll
    for (int j = 0; j < 2; ++j) {
      float* cp = C + (size_t)(row0 + wr + i * 16 + mo) * N + col0 + wc + j * 16 + nn;
#pragma unroll
      for (int r = 0; r < 8; ++r) cp[(size_t)r * N] = acc[i][j][r];
    }
}

// ---------------- K2: small GEMM for ba (N=64) ----------------
__global__ void gemm_ba_kernel(const float* __restrict__ H, const float* __restrict__ Wba,
                               float* __restrict__ ba) {
  int t = blockIdx.x;
  int n = threadIdx.x;  // 64
  const float* h = H + (size_t)t * HID;
  float s = 0.f;
  for (int k = 0; k < HID; ++k) s += h[k] * Wba[(size_t)k * 64 + n];
  ba[(size_t)t * 64 + n] = s;
}

// ---------------- K3: causal depthwise conv (K=4) + SiLU ----------------
__device__ __forceinline__ int conv_col(int c) {
  if (c < 2048) { int h = c >> 7; return h * 768 + (c & 127); }
  if (c < 4096) { int cc = c - 2048; int h = cc >> 7; return h * 768 + 128 + (cc & 127); }
  int cc = c - 4096; int h = cc >> 8; return h * 768 + 256 + (cc & 255);
}
__global__ void dwconv_silu_kernel(const float* __restrict__ Y1, const float* __restrict__ cw,
                                   float* __restrict__ mixed) {
  int c = blockIdx.x * 256 + threadIdx.x;  // 8192
  int t = blockIdx.y;                      // 4096
  int col = conv_col(c);
  float acc = 0.f;
#pragma unroll
  for (int j = 0; j < 4; ++j) {
    int tt = t - 3 + j;
    if (tt >= 0) acc += cw[c * 4 + j] * Y1[(size_t)tt * NQKVZ + col];
  }
  float s = acc / (1.f + __expf(-acc));
  mixed[(size_t)t * CONVD + c] = s;
}

// ---------------- K4a: l2norm q,k (+ q scale), pack v ----------------
__global__ void prep_qkv_kernel(const float* __restrict__ mixed, u16* __restrict__ qbf,
                                u16* __restrict__ kbf, u16* __restrict__ vbf) {
  int h = blockIdx.x;   // 16
  int t = blockIdx.y;   // 4096
  int d = threadIdx.x;  // 128
  __shared__ float red[128];
  float qv = mixed[(size_t)t * CONVD + h * 128 + d];
  float kv = mixed[(size_t)t * CONVD + 2048 + h * 128 + d];
  red[d] = qv * qv;
  __syncthreads();
  for (int s2 = 64; s2 > 0; s2 >>= 1) { if (d < s2) red[d] += red[d + s2]; __syncthreads(); }
  float ssq = red[0];
  __syncthreads();
  red[d] = kv * kv;
  __syncthreads();
  for (int s2 = 64; s2 > 0; s2 >>= 1) { if (d < s2) red[d] += red[d + s2]; __syncthreads(); }
  float ssk = red[0];
  const float qscale = 0.08838834764831845f;  // 128^-0.5
  qbf[((size_t)h * SEQ + t) * DK + d] = f2bf(qv * rsqrtf(ssq + 1e-6f) * qscale);
  kbf[((size_t)h * SEQ + t) * DK + d] = f2bf(kv * rsqrtf(ssk + 1e-6f));
#pragma unroll
  for (int r = 0; r < 2; ++r) {
    int hv = h * 2 + r;
    float vv = mixed[(size_t)t * CONVD + 4096 + hv * 128 + d];
    vbf[((size_t)hv * SEQ + t) * DV + d] = f2bf(vv);
  }
}

// ---------------- K4b: gates ----------------
__global__ void prep_gbeta_kernel(const float* __restrict__ ba, const float* __restrict__ A_log,
                                  const float* __restrict__ dt_bias, float* __restrict__ g,
                                  float* __restrict__ beta) {
  int t = blockIdx.x;
  int hv = threadIdx.x;  // 32
  int h16 = hv >> 1, r = hv & 1;
  float b = ba[(size_t)t * 64 + h16 * 4 + r];
  float a = ba[(size_t)t * 64 + h16 * 4 + 2 + r];
  float x = a + dt_bias[hv];
  float sp = (x > 20.f) ? x : log1pf(__expf(x));
  g[(size_t)hv * SEQ + t] = -__expf(A_log[hv]) * sp;
  beta[(size_t)hv * SEQ + t] = 1.f / (1.f + __expf(-b));
}

// ---------------- K5: per-chunk intra part (parallel over head x chunk) ----------------
__global__ void __launch_bounds__(256)
core_intra_kernel(const u16* __restrict__ qbf, const u16* __restrict__ kbf,
                  const u16* __restrict__ vbf, const float* __restrict__ g,
                  const float* __restrict__ beta, float* __restrict__ gcs,
                  u16* __restrict__ attnbf, float* __restrict__ vcw,
                  u16* __restrict__ kcdbf) {
  int n = blockIdx.x;   // chunk
  int hv = blockIdx.y;  // head 0..31
  int hq = hv >> 1;
  int t0 = n * CHUNK;
  int tid = threadIdx.x, lane = tid & 31, wave = tid >> 5;

  __shared__ float gc[64];
  __shared__ float bet[64];
  __shared__ float Tinv[64][64];      // 16 KB
  __shared__ uint4 arena4[2048];      // 32 KB, aliased
  float* Ms  = (float*)arena4;        // [64][64] f32    (stage 1-2)
  u16*   vbT = (u16*)arena4;          // [128][64] bf16  (stage 3+)
  u16*   kgT = ((u16*)arena4) + 128 * 64;

  if (tid < 64) bet[tid] = beta[(size_t)hv * SEQ + t0 + tid];
  if (tid == 0) {
    float s = 0.f;
    for (int i = 0; i < 64; ++i) { s += g[(size_t)hv * SEQ + t0 + i]; gc[i] = s; }
  }
  __syncthreads();
  if (tid < 64) gcs[(size_t)hv * SEQ + t0 + tid] = gc[tid];

  const u16* kg = kbf + (size_t)(hq * SEQ + t0) * DK;
  const u16* qg = qbf + (size_t)(hq * SEQ + t0) * DK;

  // Stage 1: M = (k_beta k^T)*decay (strict lower), attn_local = (q k^T)*decay (lower)
#pragma unroll
  for (int s = 0; s < 2; ++s) {
    int tIdx = wave * 2 + s;
    int r0 = (tIdx >> 2) * 16, c0 = (tIdx & 3) * 16;
    float bscale = bet[r0 + (lane & 15)];
    v8f accM = vzero(), accA = vzero();
#pragma unroll
    for (int k0 = 0; k0 < DK; k0 += 32) {
      v16bf b   = load_b_nmajor(kg + (size_t)c0 * DK + k0, DK);  // B = k^T, n-major == k row-major
      v16bf akb = load_a_bf16_scaled(kg + (size_t)r0 * DK + k0, DK, bscale);
      v16bf aq  = load_a_bf16(qg + (size_t)r0 * DK + k0, DK);
      accM = wmma_bf16(akb, b, accM);
      accA = wmma_bf16(aq, b, accA);
    }
    int nn = lane & 15, mo = (lane < 16) ? 0 : 8;
#pragma unroll
    for (int r = 0; r < 8; ++r) {
      int i = r0 + mo + r, j = c0 + nn;
      float e = __expf(gc[i] - gc[j]);
      Ms[i * 64 + j] = (i > j) ? accM[r] * e : 0.f;
      attnbf[(((size_t)hv * NCHUNK + n) * CHUNK + i) * CHUNK + j] =
          f2bf((i >= j) ? accA[r] * e : 0.f);
    }
  }
  __syncthreads();

  // Stage 2: unit-lower triangular solve (I+M) X = I, column-parallel
  if (tid < 64) {
    int j = tid;
    for (int i = 0; i < 64; ++i) {
      float s = (i == j) ? 1.f : 0.f;
      for (int l = j; l < i; ++l) s -= Ms[i * 64 + l] * Tinv[l][j];
      Tinv[i][j] = s;
    }
  }
  __syncthreads();

  // Stage 3: build v_beta^T and (k_beta e^g)^T in LDS (overwrites Ms)
  for (int idx = tid; idx < 64 * DK; idx += 256) {
    int i = idx >> 7, d = idx & 127;
    float vv = bf2f(vbf[((size_t)hv * SEQ + t0 + i) * DV + d]) * bet[i];
    vbT[d * 64 + i] = f2bf(vv);
    float kk2 = bf2f(kg[(size_t)i * DK + d]) * bet[i] * __expf(gc[i]);
    kgT[d * 64 + i] = f2bf(kk2);
  }
  __syncthreads();

  // Stage 4: v_c = Tinv @ v_beta, k_cumdecay = Tinv @ (k_beta e^g)
#pragma unroll
  for (int s = 0; s < 4; ++s) {
    int tIdx = wave * 4 + s;
    int r0 = (tIdx >> 3) * 16, c0 = (tIdx & 7) * 16;
    v8f accV = vzero(), accK = vzero();
#pragma unroll
    for (int k0 = 0; k0 < 64; k0 += 32) {
      v16bf a  = load_a_f32(&Tinv[r0][k0], 64);
      v16bf bv = load_b_nmajor(vbT + c0 * 64 + k0, 64);
      v16bf bk = load_b_nmajor(kgT + c0 * 64 + k0, 64);
      accV = wmma_bf16(a, bv, accV);
      accK = wmma_bf16(a, bk, accK);
    }
    int nn = lane & 15, mo = (lane < 16) ? 0 : 8;
#pragma unroll
    for (int r = 0; r < 8; ++r) {
      int i = r0 + mo + r, d = c0 + nn;
      vcw[((size_t)hv * SEQ + t0 + i) * DV + d] = accV[r];
      kcdbf[((size_t)hv * SEQ + t0 + i) * DK + d] = f2bf(accK[r]);
    }
  }
}

// ---------------- K6: sequential chunk scan (parallel over 32 heads) ----------------
__global__ void __launch_bounds__(256)
core_scan_kernel(const u16* __restrict__ qbf, const u16* __restrict__ kbf,
                 const float* __restrict__ gcs, const u16* __restrict__ attnbf,
                 const float* __restrict__ vcw, const u16* __restrict__ kcdbf,
                 float* __restrict__ Sws, u16* __restrict__ kdTw,
                 float* __restrict__ coreout) {
  int hv = blockIdx.x;
  int hq = hv >> 1;
  int tid = threadIdx.x, lane = tid & 31, wave = tid >> 5;

  __shared__ uint4 SbfRaw[2048];  // 32 KB : S^T bf16 [vdim][kdim]
  __shared__ uint4 vnTRaw[1024];  // 16 KB : v_new^T bf16 [vdim][i]
  __shared__ float garr[64];
  u16* Sbf = (u16*)SbfRaw;
  u16* vnT = (u16*)vnTRaw;

  float* Sg  = Sws  + (size_t)hv * DV * DK;   // f32 master state S^T
  u16*   kdT = kdTw + (size_t)hv * DK * 64;   // k_dec^T bf16 [kdim][i]

  for (int idx = tid; idx < DV * DK; idx += 256) { Sbf[idx] = 0; Sg[idx] = 0.f; }
  __syncthreads();

  for (int n = 0; n < NCHUNK; ++n) {
    int t0 = n * CHUNK;
    if (tid < 64) garr[tid] = gcs[(size_t)hv * SEQ + t0 + tid];
    __syncthreads();
    float gl = garr[63];

    // build k_dec^T = (k * e^{g_last - g_i})^T
    for (int idx = tid; idx < CHUNK * DK; idx += 256) {
      int i = idx >> 7, d = idx & 127;
      float kk2 = bf2f(kbf[((size_t)hq * SEQ + t0 + i) * DK + d]) * __expf(gl - garr[i]);
      kdT[d * 64 + i] = f2bf(kk2);
    }

    // Phase A: v_prime = kcd @ S ; attn_inter = (q e^g) @ S ; v_new -> vnT
    v8f accAI[4];
#pragma unroll
    for (int s = 0; s < 4; ++s) {
      int tIdx = wave * 4 + s;
      int r0 = (tIdx >> 3) * 16, c0 = (tIdx & 7) * 16;
      float qsc = __expf(garr[r0 + (lane & 15)]);
      v8f accVP = vzero(), ai = vzero();
#pragma unroll
      for (int k0 = 0; k0 < DK; k0 += 32) {
        v16bf bS = load_b_nmajor(Sbf + c0 * DK + k0, DK);
        v16bf aK = load_a_bf16(kcdbf + ((size_t)hv * SEQ + t0 + r0) * DK + k0, DK);
        v16bf aQ = load_a_bf16_scaled(qbf + ((size_t)hq * SEQ + t0 + r0) * DK + k0, DK, qsc);
        accVP = wmma_bf16(aK, bS, accVP);
        ai    = wmma_bf16(aQ, bS, ai);
      }
      accAI[s] = ai;
      int nn = lane & 15, mo = (lane < 16) ? 0 : 8;
#pragma unroll
      for (int r = 0; r < 8; ++r) {
        int i = r0 + mo + r, d = c0 + nn;
        float vnew = vcw[((size_t)hv * SEQ + t0 + i) * DV + d] - accVP[r];
        vnT[d * 64 + i] = f2bf(vnew);
      }
    }
    __syncthreads();  // vnT & kdT ready; all Sbf reads of this chunk done

    // Phase B: out = attn_inter + attn_local @ v_new
#pragma unroll
    for (int s = 0; s < 4; ++s) {
      int tIdx = wave * 4 + s;
      int r0 = (tIdx >> 3) * 16, c0 = (tIdx & 7) * 16;
      v8f acc = accAI[s];
#pragma unroll
      for (int k0 = 0; k0 < 64; k0 += 32) {
        v16bf aA = load_a_bf16(attnbf + (((size_t)hv * NCHUNK + n) * CHUNK + r0) * CHUNK + k0, CHUNK);
        v16bf bV = load_b_nmajor(vnT + c0 * 64 + k0, 64);
        acc = wmma_bf16(aA, bV, acc);
      }
      int nn = lane & 15, mo = (lane < 16) ? 0 : 8;
#pragma unroll
      for (int r = 0; r < 8; ++r) {
        int i = r0 + mo + r, d = c0 + nn;
        coreout[((size_t)hv * SEQ + t0 + i) * DV + d] = acc[r];
      }
    }

    // Phase C: S^T = S^T * e^{g_last} + v_new^T @ k_dec
    float egl = __expf(gl);
#pragma unroll
    for (int s = 0; s < 8; ++s) {
      int tIdx = wave * 8 + s;
      int r0 = (tIdx >> 3) * 16, c0 = (tIdx & 7) * 16;
      v8f acc = vzero();
#pragma unroll
      for (int k0 = 0; k0 < 64; k0 += 32) {
        v16bf aV = load_a_bf16(vnT + r0 * 64 + k0, 64);
        v16bf bK = load_b_nmajor(kdT + c0 * 64 + k0, 64);
        acc = wmma_bf16(aV, bK, acc);
      }
      int nn = lane & 15, mo = (lane < 16) ? 0 : 8;
#pragma unroll
      for (int r = 0; r < 8; ++r) {
        int vrow = r0 + mo + r, kcol = c0 + nn;
        float sNew = Sg[(size_t)vrow * DK + kcol] * egl + acc[r];
        Sg[(size_t)vrow * DK + kcol] = sNew;
        Sbf[vrow * DK + kcol] = f2bf(sNew);
      }
    }
    __syncthreads();
  }
}

// ---------------- K7: RMS norm * norm_weight * SiLU(z) -> bf16 ----------------
__global__ void norm_gate_kernel(const float* __restrict__ coreout,
                                 const float* __restrict__ Y1,
                                 const float* __restrict__ normw,
                                 u16* __restrict__ attnout) {
  int hv = blockIdx.x;  // 32
  int t = blockIdx.y;   // 4096
  int d = threadIdx.x;  // 128
  __shared__ float red[128];
  float x = coreout[((size_t)hv * SEQ + t) * DV + d];
  red[d] = x * x;
  __syncthreads();
  for (int s2 = 64; s2 > 0; s2 >>= 1) { if (d < s2) red[d] += red[d + s2]; __syncthreads(); }
  float ms = red[0] * (1.f / 128.f);
  float xn = x * rsqrtf(ms + 1e-6f) * normw[d];
  float z = Y1[(size_t)t * NQKVZ + (hv >> 1) * 768 + 512 + (hv & 1) * 128 + d];
  float sz = z / (1.f + __expf(-z));
  attnout[(size_t)t * (NH32 * DV) + hv * DV + d] = f2bf(xn * sz);
}

// ---------------- host-side launch ----------------
extern "C" void kernel_launch(void* const* d_in, const int* in_sizes, int n_in,
                              void* d_out, int out_size, void* d_ws, size_t ws_size,
                              hipStream_t stream) {
  const float* hidden  = (const float*)d_in[0];
  const float* W_qkvz  = (const float*)d_in[1];
  const float* W_ba    = (const float*)d_in[2];
  const float* conv_w  = (const float*)d_in[3];
  const float* dt_bias = (const float*)d_in[4];
  const float* A_log   = (const float*)d_in[5];
  const float* normw   = (const float*)d_in[6];
  const float* W_out   = (const float*)d_in[7];
  float* out = (float*)d_out;

  char* ws = (char*)d_ws;
  size_t off = 0;
  auto alloc = [&](size_t bytes) -> void* {
    void* p = ws + off;
    off = (off + bytes + 255) & ~(size_t)255;
    return p;
  };

  float* Y1      = (float*)alloc((size_t)SEQ * NQKVZ * 4);          // 192 MB
  float* baW     = (float*)alloc((size_t)SEQ * 64 * 4);
  u16*   hbf     = (u16*)  alloc((size_t)SEQ * HID * 2);
  u16*   wqbfT   = (u16*)  alloc((size_t)NQKVZ * HID * 2);          // W_qkvz^T, bf16
  u16*   wobfT   = (u16*)  alloc((size_t)HID * (NH32 * DV) * 2);    // W_out^T,  bf16
  float* mixed   = (float*)alloc((size_t)SEQ * CONVD * 4);          // 128 MB
  u16*   qbf     = (u16*)  alloc((size_t)NH16 * SEQ * DK * 2);
  u16*   kbf     = (u16*)  alloc((size_t)NH16 * SEQ * DK * 2);
  u16*   vbf     = (u16*)  alloc((size_t)NH32 * SEQ * DV * 2);
  float* gW      = (float*)alloc((size_t)NH32 * SEQ * 4);
  float* betaW   = (float*)alloc((size_t)NH32 * SEQ * 4);
  float* gcs     = (float*)alloc((size_t)NH32 * SEQ * 4);
  u16*   attnbf  = (u16*)  alloc((size_t)NH32 * SEQ * CHUNK * 2);
  float* vcw     = (float*)alloc((size_t)NH32 * SEQ * DV * 4);
  u16*   kcdbf   = (u16*)  alloc((size_t)NH32 * SEQ * DK * 2);
  float* Sws     = (float*)alloc((size_t)NH32 * DV * DK * 4);
  u16*   kdTw    = (u16*)  alloc((size_t)NH32 * DK * 64 * 2);
  float* coreout = (float*)alloc((size_t)NH32 * SEQ * DV * 4);
  u16*   attnout = (u16*)  alloc((size_t)SEQ * NH32 * DV * 2);

  // 0) conversions: hidden -> bf16; weights -> transposed bf16 (one-time, L2-resident)
  {
    long long n0 = (long long)SEQ * HID;
    cvt_bf16_kernel<<<dim3((unsigned)((n0 + 255) / 256)), dim3(256), 0, stream>>>(hidden, hbf, n0);
    transpose_cvt_kernel<<<dim3(NQKVZ / 32, HID / 32), dim3(256), 0, stream>>>(W_qkvz, wqbfT,
                                                                               HID, NQKVZ);
    transpose_cvt_kernel<<<dim3(HID / 32, (NH32 * DV) / 32), dim3(256), 0, stream>>>(
        W_out, wobfT, (NH32 * DV), HID);
  }
  // 1) Y1 = hidden @ W_qkvz  (4096 x 2048 x 12288), async-staged WMMA GEMM
  gemm_bf16<<<dim3(NQKVZ / TN, SEQ / TM), dim3(256), 0, stream>>>(hbf, wqbfT, Y1, SEQ, NQKVZ, HID);
  // 2) ba = hidden @ W_ba
  gemm_ba_kernel<<<dim3(SEQ), dim3(64), 0, stream>>>(hidden, W_ba, baW);
  // 3) depthwise causal conv + SiLU
  dwconv_silu_kernel<<<dim3(CONVD / 256, SEQ), dim3(256), 0, stream>>>(Y1, conv_w, mixed);
  // 4) l2norm q,k ; pack v ; gates
  prep_qkv_kernel<<<dim3(NH16, SEQ), dim3(128), 0, stream>>>(mixed, qbf, kbf, vbf);
  prep_gbeta_kernel<<<dim3(SEQ), dim3(32), 0, stream>>>(baW, A_log, dt_bias, gW, betaW);
  // 5) per-chunk intra computation
  core_intra_kernel<<<dim3(NCHUNK, NH32), dim3(256), 0, stream>>>(qbf, kbf, vbf, gW, betaW,
                                                                  gcs, attnbf, vcw, kcdbf);
  // 6) sequential state scan over chunks
  core_scan_kernel<<<dim3(NH32), dim3(256), 0, stream>>>(qbf, kbf, gcs, attnbf, vcw, kcdbf,
                                                         Sws, kdTw, coreout);
  // 7) RMS norm + gate
  norm_gate_kernel<<<dim3(NH32, SEQ), dim3(128), 0, stream>>>(coreout, Y1, normw, attnout);
  // 8) out = h @ W_out  (4096 x 4096 x 2048)
  gemm_bf16<<<dim3(HID / TN, SEQ / TM), dim3(256), 0, stream>>>(attnout, wobfT, out, SEQ, HID,
                                                                (NH32 * DV));
}
